// DeepSeekMoEWithSharedExpert_23433341567241
// MI455X (gfx1250) — matrix operand order repaired and checked by reading, not compiled, not run on hardware
//
#include <hip/hip_runtime.h>

// ---------------- problem constants ----------------
#define BDIM 4
#define SDIM 2048
#define DDIM 2048
#define FDIM 1408
#define EDIM 8
#define TTOT (BDIM * SDIM)        // 8192 tokens
#define KT_UP (DDIM / 32)         // 64 K-tiles for up proj
#define NT_UP (FDIM / 16)         // 88 N-tiles for up proj
#define KT_DN (FDIM / 32)         // 44 K-tiles for down proj
#define NT_DN (DDIM / 16)         // 128 N-tiles for down proj

typedef __attribute__((ext_vector_type(16))) __bf16 v16bf;
typedef __attribute__((ext_vector_type(8)))  __bf16 v8bf;
typedef __attribute__((ext_vector_type(8)))  float  v8f;
typedef __attribute__((ext_vector_type(4)))  unsigned int u32x4;
typedef __attribute__((ext_vector_type(8)))  int          i32x8;
typedef __attribute__((ext_vector_type(4)))  int          i32x4;

// ---------------- fragment load helpers ----------------
// A-matrix 16x32 bf16 fragment (ISA 7.12.2): row m = lane&15,
// lanes 0-15 hold K=0..7 & 16..23, lanes 16-31 hold K=8..15 & 24..31.
// p points at row base + half*8; loads [p,p+8) and [p+16,p+24).
__device__ __forceinline__ v16bf load_a_frag(const __bf16* p) {
    v8bf lo = *(const v8bf*)(p);
    v8bf hi = *(const v8bf*)(p + 16);
    v16bf r;
#pragma unroll
    for (int i = 0; i < 8; ++i) { r[i] = lo[i]; r[i + 8] = hi[i]; }
    return r;
}

// B-matrix 32x16 fragment: weights are pre-swizzled so each lane's 16
// bf16 values are contiguous (32B) -> single vector load.
__device__ __forceinline__ v16bf load_b_frag(const __bf16* p) {
    return *(const v16bf*)(p);
}

__device__ __forceinline__ v8f wmma_bf16(v16bf a, v16bf b, v8f c) {
    return __builtin_amdgcn_wmma_f32_16x16x32_bf16(
        false, a, false, b, (short)0, c, false, false);
}

// ---------------- kernel 1: x fp32 -> bf16 ----------------
__global__ void convert_x_kernel(const float* __restrict__ x, __bf16* __restrict__ xb) {
    size_t i = ((size_t)blockIdx.x * blockDim.x + threadIdx.x) * 4;
    float4 v = *(const float4*)(x + i);
    xb[i + 0] = (__bf16)v.x;
    xb[i + 1] = (__bf16)v.y;
    xb[i + 2] = (__bf16)v.z;
    xb[i + 3] = (__bf16)v.w;
}

// ---------------- kernel 2: weight fp32 [K,N] -> bf16 WMMA-tiled ----------------
// Tile = 32(K) x 16(N). Element (k,n): lane = (n&15) + ((k&8)<<1), pos = (k&7) + ((k&16)>>1).
// Tile order: [ntile][ktile] so the GEMM K-loop streams contiguously.
__global__ void convert_w_tiled_kernel(const float* __restrict__ src,
                                       __bf16* __restrict__ dst,
                                       int K, int N) {
    int mat = blockIdx.y;
    size_t nelem = (size_t)K * N;
    size_t gid = (size_t)blockIdx.x * blockDim.x + threadIdx.x;
    if (gid >= nelem) return;
    const float* s = src + (size_t)mat * nelem;
    int KT = K >> 5;
    __bf16* d = dst + (size_t)mat * nelem;
    int k = (int)(gid / N);
    int n = (int)(gid % N);
    int kt = k >> 5, nt = n >> 4;
    int lane = (n & 15) + ((k & 8) << 1);
    int pos = (k & 7) + ((k & 16) >> 1);
    size_t off = (((size_t)nt * KT + kt) << 9) + (size_t)lane * 16 + pos;
    d[off] = (__bf16)s[gid];
}

// ---------------- kernel 3: router (one wave32 per token) ----------------
__global__ void router_kernel(const float* __restrict__ x,
                              const float* __restrict__ gate_w,
                              int* __restrict__ counts,
                              int* __restrict__ idx_lists,
                              float* __restrict__ wgt_lists) {
    int t = blockIdx.x * 8 + (threadIdx.x >> 5);
    int lane = threadIdx.x & 31;
    const float* xr = x + (size_t)t * DDIM;
    float acc[EDIM];
#pragma unroll
    for (int e = 0; e < EDIM; ++e) acc[e] = 0.f;
    for (int k = lane; k < DDIM; k += 32) {
        float xv = xr[k];
        const float* g = gate_w + (size_t)k * EDIM;
#pragma unroll
        for (int e = 0; e < EDIM; ++e) acc[e] += xv * g[e];
    }
#pragma unroll
    for (int e = 0; e < EDIM; ++e) {
#pragma unroll
        for (int m = 16; m >= 1; m >>= 1) acc[e] += __shfl_xor(acc[e], m, 32);
    }
    if (lane == 0) {
        float mx = acc[0];
#pragma unroll
        for (int e = 1; e < EDIM; ++e) mx = fmaxf(mx, acc[e]);
        float p[EDIM];
#pragma unroll
        for (int e = 0; e < EDIM; ++e) p[e] = __expf(acc[e] - mx);
        int e1 = 0;
#pragma unroll
        for (int e = 1; e < EDIM; ++e) if (p[e] > p[e1]) e1 = e;
        int e2 = (e1 == 0) ? 1 : 0;
#pragma unroll
        for (int e = 0; e < EDIM; ++e) if (e != e1 && p[e] > p[e2]) e2 = e;
        float inv = 1.f / (p[e1] + p[e2]);
        int s1 = atomicAdd(&counts[e1], 1);
        idx_lists[(size_t)e1 * TTOT + s1] = t;
        wgt_lists[(size_t)e1 * TTOT + s1] = p[e1] * inv;
        int s2 = atomicAdd(&counts[e2], 1);
        idx_lists[(size_t)e2 * TTOT + s2] = t;
        wgt_lists[(size_t)e2 * TTOT + s2] = p[e2] * inv;
    }
}

// ---------------- kernel 4: fused SwiGLU expert (TDM gather -> up -> LDS -> down) ----
// One block = 16 gathered tokens; 8 waves split N-tiles. The 16-row token
// gather is done by the Tensor Data Mover in gather mode (16-bit row indices),
// DMA'ing the A strip [16 x D] bf16 straight into LDS.
// idx==nullptr -> shared expert (identity gather, weight 1, plain store).
__global__ __launch_bounds__(256) void moe_expert_kernel(
    const __bf16* __restrict__ xb,
    const __bf16* __restrict__ w1t,
    const __bf16* __restrict__ w3t,
    const __bf16* __restrict__ w2t,
    const int* __restrict__ idx,
    const float* __restrict__ wgt,
    const int* __restrict__ count,
    float* __restrict__ out) {

    if (idx) {
        int c = *count;
        if ((int)blockIdx.x * 16 >= c) return;
    }

    __shared__ alignas(16) __bf16 As[16 * DDIM];   // 64 KB gathered A strip
    __shared__ alignas(16) __bf16 Hs[16 * FDIM];   // 45 KB hidden strip
    __shared__ int   toks[16];
    __shared__ float wrow[16];

    int tid = threadIdx.x;
    if (tid < 16) {
        int slot = blockIdx.x * 16 + tid;
        if (idx) {
            int c = *count;
            if (slot < c) { toks[tid] = idx[slot]; wrow[tid] = wgt[slot]; }
            else          { toks[tid] = 0;         wrow[tid] = 0.f; }
        } else {
            toks[tid] = slot;
            wrow[tid] = 1.f;
        }
    }
    __syncthreads();

    int lane = tid & 31;
    int wid  = tid >> 5;
    int l    = lane & 15;
    int half = lane >> 4;

    // ---- TDM gather: DMA 16 token rows of xb into As (wave 0 issues) ----
    if (wid == 0) {
        unsigned long long gaddr = (unsigned long long)(const void*)xb;
        unsigned int lds_off =
            (unsigned int)__builtin_amdgcn_readfirstlane((int)(unsigned int)(unsigned long long)(uintptr_t)As);

        // D# group 0: count=1, gather_mode=1 (bit31), 16-bit indices (bit30=0),
        // lds_addr, 57-bit global_addr, type=2 (bits 127:126).
        u32x4 g0;
        g0[0] = 0x80000001u;
        g0[1] = lds_off;
        g0[2] = (unsigned int)__builtin_amdgcn_readfirstlane((int)(unsigned int)(gaddr & 0xffffffffu));
        g0[3] = (unsigned int)__builtin_amdgcn_readfirstlane(
                    (int)(((unsigned int)(gaddr >> 32) & 0x01ffffffu) | 0x80000000u));

        // D# group 1: data_size=1 (2B) at bits 17:16; tensor_dim0=D (bits 79:48);
        // tensor_dim1=16 (bits 111:80); tile_dim0=D (bits 127:112);
        // tile_dim1=#indices=16 (bits 143:128); tensor_dim0_stride=D (bits 207:160).
        i32x8 g1;
        g1[0] = 0x00010000;                         // data_size = 2 bytes
        g1[1] = (int)((DDIM & 0xffff) << 16);       // tensor_dim0 lo16 in [63:48]
        g1[2] = (int)((DDIM >> 16) | (16 << 16));   // tensor_dim0 hi16, tensor_dim1 lo16
        g1[3] = (int)((DDIM & 0xffff) << 16);       // tile_dim0 in [127:112]
        g1[4] = 16;                                 // tile_dim1 = 16 valid indices
        g1[5] = DDIM;                               // tensor_dim0_stride lo32
        g1[6] = 0;
        g1[7] = 0;

        // D# groups 2-3: sixteen 16-bit row indices (gathered token ids).
        i32x4 g2, g3;
#pragma unroll
        for (int i = 0; i < 4; ++i) {
            g2[i] = __builtin_amdgcn_readfirstlane(
                (toks[2 * i] & 0xffff) | (toks[2 * i + 1] << 16));
            g3[i] = __builtin_amdgcn_readfirstlane(
                (toks[8 + 2 * i] & 0xffff) | (toks[8 + 2 * i + 1] << 16));
        }

        i32x8 gx = {0, 0, 0, 0, 0, 0, 0, 0};   // unused trailing descriptor group

        __builtin_amdgcn_tensor_load_to_lds(g0, g1, g2, g3, gx, 0);
        __builtin_amdgcn_s_wait_tensorcnt(0);
    }
    __syncthreads();

    const __bf16* arow = &As[l * DDIM + half * 8];

    // ---- up: H = silu(X@W1) * (X@W3), H strip kept in LDS ----
    for (int nf = wid; nf < NT_UP; nf += 8) {
        v8f acc1 = {};
        v8f acc3 = {};
        const __bf16* b1p = w1t + ((size_t)nf * KT_UP << 9) + lane * 16;
        const __bf16* b3p = w3t + ((size_t)nf * KT_UP << 9) + lane * 16;
        for (int kt = 0; kt < KT_UP; ++kt) {
            v16bf a  = load_a_frag(arow + kt * 32);
            v16bf b1 = load_b_frag(b1p + (size_t)kt * 512);
            v16bf b3 = load_b_frag(b3p + (size_t)kt * 512);
            acc1 = wmma_bf16(a, b1, acc1);
            acc3 = wmma_bf16(a, b3, acc3);
        }
#pragma unroll
        for (int i = 0; i < 8; ++i) {
            int row = half * 8 + i;
            float v1 = acc1[i];
            float h = (v1 / (1.f + __expf(-v1))) * acc3[i];
            Hs[row * FDIM + nf * 16 + l] = (__bf16)h;
        }
    }
    __syncthreads();

    // ---- down: O = H @ W2, scatter-add weighted rows ----
    for (int nd = wid; nd < NT_DN; nd += 8) {
        v8f acc = {};
        const __bf16* b2p = w2t + ((size_t)nd * KT_DN << 9) + lane * 16;
        const __bf16* hp  = &Hs[l * FDIM + half * 8];
        for (int kt = 0; kt < KT_DN; ++kt) {
            v16bf a = load_a_frag(hp + kt * 32);
            v16bf b = load_b_frag(b2p + (size_t)kt * 512);
            acc = wmma_bf16(a, b, acc);
        }
#pragma unroll
        for (int i = 0; i < 8; ++i) {
            int row = half * 8 + i;
            int t = toks[row];
            float w = wrow[row];
            size_t o = (size_t)t * DDIM + nd * 16 + l;
            if (idx) {
                if (w != 0.f) atomicAdd(&out[o], acc[i] * w);
            } else {
                out[o] = acc[i];     // shared expert runs first: plain store
            }
        }
    }
}

// ---------------- host launcher ----------------
extern "C" void kernel_launch(void* const* d_in, const int* in_sizes, int n_in,
                              void* d_out, int out_size, void* d_ws, size_t ws_size,
                              hipStream_t stream) {
    const float* x         = (const float*)d_in[0];
    const float* gate_w    = (const float*)d_in[1];
    const float* shared_w1 = (const float*)d_in[2];
    const float* shared_w2 = (const float*)d_in[3];
    const float* shared_w3 = (const float*)d_in[4];
    const float* expert_w1 = (const float*)d_in[5];
    const float* expert_w2 = (const float*)d_in[6];
    const float* expert_w3 = (const float*)d_in[7];
    float* out = (float*)d_out;

    char* wp = (char*)d_ws;
    auto carve = [&](size_t bytes) -> void* {
        void* p = (void*)wp;
        wp += (bytes + 255) & ~(size_t)255;
        return p;
    };
    const size_t matsz = (size_t)DDIM * FDIM;           // elements per proj matrix
    __bf16* xb   = (__bf16*)carve((size_t)TTOT * DDIM * 2);
    __bf16* sw1t = (__bf16*)carve(matsz * 2);
    __bf16* sw3t = (__bf16*)carve(matsz * 2);
    __bf16* sw2t = (__bf16*)carve(matsz * 2);
    __bf16* ew1t = (__bf16*)carve((size_t)EDIM * matsz * 2);
    __bf16* ew3t = (__bf16*)carve((size_t)EDIM * matsz * 2);
    __bf16* ew2t = (__bf16*)carve((size_t)EDIM * matsz * 2);
    int*    counts = (int*)carve(EDIM * sizeof(int));
    int*    idxl   = (int*)carve((size_t)EDIM * TTOT * sizeof(int));
    float*  wgtl   = (float*)carve((size_t)EDIM * TTOT * sizeof(float));

    (void)hipMemsetAsync(counts, 0, EDIM * sizeof(int), stream);

    convert_x_kernel<<<(TTOT * (size_t)DDIM) / 1024, 256, 0, stream>>>(x, xb);

    unsigned wblocks = (unsigned)((matsz + 255) / 256);
    convert_w_tiled_kernel<<<dim3(wblocks, 1), 256, 0, stream>>>(shared_w1, sw1t, DDIM, FDIM);
    convert_w_tiled_kernel<<<dim3(wblocks, 1), 256, 0, stream>>>(shared_w3, sw3t, DDIM, FDIM);
    convert_w_tiled_kernel<<<dim3(wblocks, 1), 256, 0, stream>>>(shared_w2, sw2t, FDIM, DDIM);
    convert_w_tiled_kernel<<<dim3(wblocks, EDIM), 256, 0, stream>>>(expert_w1, ew1t, DDIM, FDIM);
    convert_w_tiled_kernel<<<dim3(wblocks, EDIM), 256, 0, stream>>>(expert_w3, ew3t, DDIM, FDIM);
    convert_w_tiled_kernel<<<dim3(wblocks, EDIM), 256, 0, stream>>>(expert_w2, ew2t, FDIM, DDIM);

    router_kernel<<<TTOT / 8, 256, 0, stream>>>(x, gate_w, counts, idxl, wgtl);

    // shared expert: identity gather, plain stores (initializes out)
    moe_expert_kernel<<<TTOT / 16, 256, 0, stream>>>(
        xb, sw1t, sw3t, sw2t, nullptr, nullptr, nullptr, out);

    // routed experts: gathered tokens, weighted atomic accumulation
    for (int e = 0; e < EDIM; ++e) {
        moe_expert_kernel<<<TTOT / 16, 256, 0, stream>>>(
            xb,
            ew1t + (size_t)e * matsz,
            ew3t + (size_t)e * matsz,
            ew2t + (size_t)e * matsz,
            idxl + (size_t)e * TTOT,
            wgtl + (size_t)e * TTOT,
            counts + e,
            out);
    }
}
